// CausalSelfAttention_58119497449725
// MI455X (gfx1250) — matrix operand order repaired and checked by reading, not compiled, not run on hardware
//
#include <hip/hip_runtime.h>
#include <hip/hip_bf16.h>

// ---------------------------------------------------------------------------
// CausalSelfAttention for MI455X (gfx1250): f16 WMMA pipeline, f32 accumulate.
// Double-buffered Tensor-Data-Mover (tensor_load_to_lds) staging of the shared
// operands (weight tiles in the GEMMs, K/V blocks in flash attention).
// B=2, N=2048, D=2048, H=16, hd=128, inner=2048.
// ---------------------------------------------------------------------------

typedef __attribute__((ext_vector_type(16))) _Float16 v16h;
typedef __attribute__((ext_vector_type(8)))  _Float16 v8h;
typedef __attribute__((ext_vector_type(4)))  _Float16 v4h;
typedef __attribute__((ext_vector_type(8)))  float    v8f;
typedef __attribute__((ext_vector_type(4)))  unsigned int u32x4;
typedef __attribute__((ext_vector_type(8)))  int      i32x8;
typedef __attribute__((ext_vector_type(4)))  int      i32x4;

#define Bc 2
#define Nc 2048
#define Dc 2048
#define Hc 16
#define HDc 128
#define INNERc 2048
#define TRIPLEc 6144

#if defined(__has_builtin)
#if __has_builtin(__builtin_amdgcn_tensor_load_to_lds) && __has_builtin(__builtin_amdgcn_s_wait_tensorcnt)
#define HAVE_TDM 1
#endif
#endif

__device__ __forceinline__ v8f wmma_f16(v16h a, v16h b, v8f c) {
    return __builtin_amdgcn_wmma_f32_16x16x32_f16(
        /*neg_a=*/false, a, /*neg_b=*/false, b,
        /*c_mod=*/(short)0, c, /*reuse_a=*/false, /*reuse_b=*/false);
}

__device__ __forceinline__ v16h mk16(v8h lo, v8h hi) {
    v16h r;
#pragma unroll
    for (int i = 0; i < 8; ++i) { r[i] = lo[i]; r[i + 8] = hi[i]; }
    return r;
}

// A operand, 16x32 f16 tile, row-major source with leading dim `ld`.
__device__ __forceinline__ v16h ldA(const _Float16* __restrict__ base, size_t ld,
                                    int m0, int k0, int lane) {
    const _Float16* p = base + (size_t)(m0 + (lane & 15)) * ld + k0 + ((lane >> 4) << 3);
    return mk16(*(const v8h*)p, *(const v8h*)(p + 16));
}

// B operand, 32x16 f16 tile from a K-major [K][Ncols] source.
__device__ __forceinline__ v16h ldB(const _Float16* __restrict__ base, size_t ld,
                                    int k0, int n0, int lane) {
    const _Float16* p = base + (size_t)(k0 + ((lane >> 4) << 3)) * ld + n0 + (lane & 15);
    v16h r;
#pragma unroll
    for (int i = 0; i < 8; ++i) { r[i] = p[(size_t)i * ld]; r[i + 8] = p[(size_t)(i + 16) * ld]; }
    return r;
}

#ifdef HAVE_TDM
typedef __attribute__((address_space(3))) _Float16 lds_f16;
__device__ __forceinline__ unsigned lds_offset(const _Float16* p) {
    return (unsigned)(unsigned long long)(const lds_f16*)p;
}

// Build a D# (groups 0/1) and issue TENSOR_LOAD_TO_LDS for a 2-D f16 tile:
// tile_d0 contiguous elements per row, tile_d1 rows, row stride `stride0`
// (elements); 1-D copies use tile_d1 = 0.
__device__ __forceinline__ void tdm_load(unsigned lds_off, const void* gaddr,
                                         unsigned tile_d0, unsigned tile_d1,
                                         unsigned tensor_d0, unsigned tensor_d1,
                                         unsigned long long stride0) {
    unsigned long long ga = (unsigned long long)gaddr;
    u32x4 g0;
    g0[0] = 1u;                                   // count=1 (valid user descriptor)
    g0[1] = lds_off;                              // lds_addr (bytes)
    g0[2] = (unsigned)ga;                         // global_addr[31:0]
    g0[3] = (unsigned)(ga >> 32) | (2u << 30);    // global_addr[56:32] | type=2
    unsigned w0 = (1u << 16);                     // data_size=1 -> 2 bytes
    unsigned w1 = (tensor_d0 & 0xFFFFu) << 16;    // tensor_dim0[15:0]
    unsigned w2 = (tensor_d0 >> 16) | ((tensor_d1 & 0xFFFFu) << 16);
    unsigned w3 = (tensor_d1 >> 16) | (tile_d0 << 16);   // tile_dim0
    unsigned w4 = (tile_d1 & 0xFFFFu);            // tile_dim1 (tile_dim2 = 0)
    unsigned w5 = (unsigned)stride0;              // tensor_dim0_stride[31:0]
    unsigned w6 = (unsigned)(stride0 >> 32) & 0xFFFFu;
    unsigned w7 = 0;
    i32x8 g1 = { (int)w0, (int)w1, (int)w2, (int)w3,
                 (int)w4, (int)w5, (int)w6, (int)w7 };
    i32x4 z4 = { 0, 0, 0, 0 };
#if __clang_major__ >= 23
    i32x8 z8 = { 0, 0, 0, 0, 0, 0, 0, 0 };
    __builtin_amdgcn_tensor_load_to_lds(g0, g1, z4, z4, z8, 0);
#else
    __builtin_amdgcn_tensor_load_to_lds(g0, g1, z4, z4, 0);
#endif
}
#endif // HAVE_TDM

// ---------------------------------------------------------------------------
// fp32 -> f16 convert (x)
// ---------------------------------------------------------------------------
__global__ void cvt_f32_f16(const float* __restrict__ in, _Float16* __restrict__ out, int n4) {
    int i = blockIdx.x * blockDim.x + threadIdx.x;
    if (i < n4) {
        float4 v = ((const float4*)in)[i];
        v4h o = { (_Float16)v.x, (_Float16)v.y, (_Float16)v.z, (_Float16)v.w };
        ((v4h*)out)[i] = o;
    }
}

// ---------------------------------------------------------------------------
// LDS-tiled transpose + convert: in[R][C] f32 -> out[C][R] f16
// ---------------------------------------------------------------------------
__global__ void transpose_f32_f16(const float* __restrict__ in, _Float16* __restrict__ out,
                                  int R, int C) {
    __shared__ float tile[32][33];
    int r0 = blockIdx.y * 32, c0 = blockIdx.x * 32;
#pragma unroll
    for (int i = threadIdx.y; i < 32; i += 8)
        tile[i][threadIdx.x] = in[(size_t)(r0 + i) * C + c0 + threadIdx.x];
    __syncthreads();
#pragma unroll
    for (int i = threadIdx.y; i < 32; i += 8)
        out[(size_t)(c0 + i) * R + r0 + threadIdx.x] = (_Float16)tile[threadIdx.x][i];
}

// ---------------------------------------------------------------------------
// GEMM mainloop pattern: block tile 64(M) x 256(N), 8 waves (2M x 4N),
// wave tile 32x64 (8 accumulators). B tile (32x256 = 16KB) staged into LDS
// by the TDM, double-buffered; A direct from global with prefetch.
// ---------------------------------------------------------------------------
#define GEMM_MAINLOOP(ABASE, ALD, BBASE, BLD, KDIM)                                   \
    const int lane = threadIdx.x & 31, wv = threadIdx.x >> 5;                          \
    const int m0 = blockIdx.x * 64 + (wv & 1) * 32;                                    \
    const int nblk = blockIdx.y * 256;                                                 \
    const int nloc = (wv >> 1) * 64;                                                   \
    const int n0 = nblk + nloc;                                                        \
    v8f acc[2][4] = {};                                                                \
    IF_TDM(__shared__ alignas(16) _Float16 bbuf[2][32 * 256];                          \
           if (wv == 0)                                                                \
               tdm_load(lds_offset(&bbuf[0][0]), (BBASE) + nblk,                       \
                        256, 32, (BLD), (KDIM), (BLD));)                               \
    for (int k0 = 0; k0 < (KDIM); k0 += 32) {                                          \
        IF_TDM(                                                                        \
            const int cur = (k0 >> 5) & 1;                                             \
            if (wv == 0) {                                                             \
                if (k0 + 32 < (KDIM)) {                                                \
                    tdm_load(lds_offset(&bbuf[cur ^ 1][0]),                            \
                             (BBASE) + (size_t)(k0 + 32) * (BLD) + nblk,               \
                             256, 32, (BLD), (KDIM), (BLD));                           \
                    __builtin_amdgcn_s_wait_tensorcnt(2);                              \
                } else {                                                               \
                    __builtin_amdgcn_s_wait_tensorcnt(0);                              \
                }                                                                      \
            }                                                                          \
            __syncthreads();                                                           \
            asm volatile("" ::: "memory");)                                            \
        v16h a0 = ldA((ABASE), (ALD), m0, k0, lane);                                   \
        v16h a1 = ldA((ABASE), (ALD), m0 + 16, k0, lane);                              \
        if (k0 + 32 < (KDIM))                                                          \
            __builtin_prefetch((ABASE) + (size_t)(m0 + (lane & 15)) * (ALD) + k0 + 32, \
                               0, 3);                                                  \
        _Pragma("unroll")                                                              \
        for (int t = 0; t < 4; ++t) {                                                  \
            IF_TDM(const _Float16* bp = &bbuf[cur][0] + ((lane >> 4) << 3) * 256 + nloc + 16 * t + (lane & 15); \
                   v16h b;                                                             \
                   for (int i = 0; i < 8; ++i) { b[i] = bp[i * 256]; b[i + 8] = bp[(i + 16) * 256]; })  \
            NO_TDM(v16h b = ldB((BBASE), (BLD), k0, n0 + 16 * t, lane);)               \
            acc[0][t] = wmma_f16(a0, b, acc[0][t]);                                    \
            acc[1][t] = wmma_f16(a1, b, acc[1][t]);                                    \
        }                                                                              \
        IF_TDM(__syncthreads();)                                                       \
    }

#ifdef HAVE_TDM
#define IF_TDM(...) __VA_ARGS__
#define NO_TDM(...)
#else
#define IF_TDM(...)
#define NO_TDM(...) __VA_ARGS__
#endif

// ---------------------------------------------------------------------------
// QKV GEMM: scatter into q/k/v buffers laid out [B,H,N,hd] as f16.
// ---------------------------------------------------------------------------
__global__ void gemm_qkv(const _Float16* __restrict__ xh, const _Float16* __restrict__ wt,
                         _Float16* __restrict__ qraw, _Float16* __restrict__ kraw,
                         _Float16* __restrict__ vh) {
    GEMM_MAINLOOP(xh, Dc, wt, TRIPLEc, Dc)

    const int hi = lane >> 4, ln = lane & 15;
#pragma unroll
    for (int t = 0; t < 4; ++t) {
        int col = n0 + 16 * t + ln;
        int sel = col >> 11;            // 0=q 1=k 2=v
        int jj = col & 2047;
        int h = jj >> 7, d = jj & 127;
        _Float16* dst = (sel == 0) ? qraw : ((sel == 1) ? kraw : vh);
#pragma unroll
        for (int u = 0; u < 2; ++u) {
#pragma unroll
            for (int r = 0; r < 8; ++r) {
                int m = m0 + 16 * u + r + 8 * hi;
                int b = m >> 11, nn = m & 2047;
                dst[(((size_t)(b * Hc + h)) * Nc + nn) * HDc + d] = (_Float16)acc[u][t][r];
            }
        }
    }
}

// ---------------------------------------------------------------------------
// Output GEMM: out[m, n] = sum_k oh[m,k] * w_o[n,k]  (fp32 result)
// ---------------------------------------------------------------------------
__global__ void gemm_out(const _Float16* __restrict__ oh, const _Float16* __restrict__ wot,
                         float* __restrict__ out) {
    GEMM_MAINLOOP(oh, INNERc, wot, Dc, INNERc)

    const int hi = lane >> 4, ln = lane & 15;
#pragma unroll
    for (int t = 0; t < 4; ++t) {
#pragma unroll
        for (int u = 0; u < 2; ++u) {
#pragma unroll
            for (int r = 0; r < 8; ++r) {
                int m = m0 + 16 * u + r + 8 * hi;
                out[(size_t)m * Dc + n0 + 16 * t + ln] = acc[u][t][r];
            }
        }
    }
}

// ---------------------------------------------------------------------------
// RoPE (rotate-half), f32 math: q in place; k transposed to [B,H,hd,N].
// ---------------------------------------------------------------------------
__global__ void rope_kernel(_Float16* __restrict__ q, const _Float16* __restrict__ kraw,
                            _Float16* __restrict__ kt) {
    int idx = blockIdx.x * blockDim.x + threadIdx.x;
    int d  = idx & 63;
    int n  = (idx >> 6) & (Nc - 1);
    int bh = idx >> 17;
    float freq = __expf(-(float)(2 * d) * (9.210340371976184f / 128.0f));
    float ang = (float)n * freq;
    float s, c;
    __sincosf(ang, &s, &c);

    size_t base = ((size_t)bh * Nc + n) * HDc;
    float q1 = (float)q[base + d], q2 = (float)q[base + d + 64];
    q[base + d]      = (_Float16)(q1 * c - q2 * s);
    q[base + d + 64] = (_Float16)(q2 * c + q1 * s);

    float k1 = (float)kraw[base + d], k2 = (float)kraw[base + d + 64];
    size_t tb = ((size_t)bh * HDc + d) * Nc + n;
    kt[tb]                   = (_Float16)(k1 * c - k2 * s);
    kt[tb + (size_t)64 * Nc] = (_Float16)(k2 * c + k1 * s);
}

// ---------------------------------------------------------------------------
// Flash attention (causal, online softmax). 4 waves/block share (b,h); one
// 16-query tile per wave. K-block (128x32 of k^T) and V-block (32x128) are
// staged once per block by the TDM, double-buffered, and consumed by all
// four waves. Waves past their causal limit skip compute (wave-uniform
// branch, EXEC stays all-1s) but keep the barrier schedule.
// ---------------------------------------------------------------------------
__global__ void flash_attn(const _Float16* __restrict__ qh, const _Float16* __restrict__ kt,
                           const _Float16* __restrict__ vh, _Float16* __restrict__ oh) {
    const int lane = threadIdx.x & 31;
    const int wv = threadIdx.x >> 5;
    const int bh = blockIdx.y;
    const int b = bh >> 4, h = bh & 15;
    const int q0 = (blockIdx.x * 4 + wv) * 16;
    const int hi = lane >> 4, ln = lane & 15;
    const float scale = 0.08838834764831845f;   // 1/sqrt(128)

#ifdef HAVE_TDM
    __shared__ alignas(16) _Float16 kbuf[2][128 * 32];   // [kappa][key], per block
    __shared__ alignas(16) _Float16 vbuf[2][32 * 128];   // [key][d], per block
#endif
    __shared__ alignas(16) _Float16 plds[4][16 * 32];

    const _Float16* qb  = qh + ((size_t)bh * Nc + q0) * HDc;
    const _Float16* ktb = kt + (size_t)bh * HDc * Nc;
    const _Float16* vb  = vh + (size_t)bh * Nc * HDc;

    // Q tile in A layout, 4 chunks of K=32 over hd=128.
    v16h aQ[4];
#pragma unroll
    for (int kc = 0; kc < 4; ++kc) {
        const _Float16* p = qb + (size_t)ln * HDc + kc * 32 + hi * 8;
        aQ[kc] = mk16(*(const v8h*)p, *(const v8h*)(p + 16));
    }

    v8f O[8] = {};
    float mrun[8], lrun[8];
#pragma unroll
    for (int r = 0; r < 8; ++r) { mrun[r] = -1e30f; lrun[r] = 0.0f; }

    const int kmax = (blockIdx.x * 4 + 3) * 16 + 15;   // block-uniform causal bound

#ifdef HAVE_TDM
    if (wv == 0) {
        tdm_load(lds_offset(&kbuf[0][0]), ktb, 32, 128, Nc, HDc, Nc);
        tdm_load(lds_offset(&vbuf[0][0]), vb, 4096, 0, 1u << 22, 1, 4096);
    }
#endif

    for (int k0 = 0; k0 <= kmax; k0 += 32) {
#ifdef HAVE_TDM
        const int cur = (k0 >> 5) & 1;
        if (wv == 0) {
            if (k0 + 32 <= kmax) {
                tdm_load(lds_offset(&kbuf[cur ^ 1][0]), ktb + k0 + 32,
                         32, 128, Nc, HDc, Nc);
                tdm_load(lds_offset(&vbuf[cur ^ 1][0]), vb + (size_t)(k0 + 32) * HDc,
                         4096, 0, 1u << 22, 1, 4096);
                __builtin_amdgcn_s_wait_tensorcnt(2);
            } else {
                __builtin_amdgcn_s_wait_tensorcnt(0);
            }
        }
        __syncthreads();
        asm volatile("" ::: "memory");
        const _Float16* kb  = &kbuf[cur][0];
        const _Float16* vbl = &vbuf[cur][0];
#endif
        if (k0 <= q0 + 15) {            // wave-uniform: EXEC all-1s inside
            // ---- S = Q K^T (two 16-key column tiles) ----
            v8f s0 = {}, s1 = {};
#pragma unroll
            for (int kc = 0; kc < 4; ++kc) {
#ifdef HAVE_TDM
                const _Float16* p = kb + (kc * 32 + hi * 8) * 32 + ln;
                v16h b0, b1;
#pragma unroll
                for (int i = 0; i < 8; ++i) {
                    b0[i] = p[i * 32];      b0[i + 8] = p[(i + 16) * 32];
                    b1[i] = p[i * 32 + 16]; b1[i + 8] = p[(i + 16) * 32 + 16];
                }
#else
                v16h b0 = ldB(ktb, Nc, kc * 32, k0, lane);
                v16h b1 = ldB(ktb, Nc, kc * 32, k0 + 16, lane);
#endif
                s0 = wmma_f16(aQ[kc], b0, s0);
                s1 = wmma_f16(aQ[kc], b1, s1);
            }

            // ---- scale + causal mask ----
            const bool needmask = (k0 + 31 > q0);
#pragma unroll
            for (int r = 0; r < 8; ++r) {
                int qrow = q0 + r + 8 * hi;
                float x0 = s0[r] * scale;
                float x1 = s1[r] * scale;
                if (needmask) {
                    if (k0 + ln > qrow)      x0 = -1e30f;
                    if (k0 + 16 + ln > qrow) x1 = -1e30f;
                }
                s0[r] = x0; s1[r] = x1;
            }

            // ---- online softmax (row reductions in 16-lane half-groups) ----
            float fs[8];
#pragma unroll
            for (int r = 0; r < 8; ++r) {
                float t = fmaxf(s0[r], s1[r]);
                t = fmaxf(t, __shfl_xor(t, 1, 32));
                t = fmaxf(t, __shfl_xor(t, 2, 32));
                t = fmaxf(t, __shfl_xor(t, 4, 32));
                t = fmaxf(t, __shfl_xor(t, 8, 32));
                float mnew = fmaxf(mrun[r], t);
                float f = __expf(mrun[r] - mnew);
                mrun[r] = mnew;
                float p0 = __expf(s0[r] - mnew);
                float p1 = __expf(s1[r] - mnew);
                float rs = p0 + p1;
                rs += __shfl_xor(rs, 1, 32);
                rs += __shfl_xor(rs, 2, 32);
                rs += __shfl_xor(rs, 4, 32);
                rs += __shfl_xor(rs, 8, 32);
                lrun[r] = lrun[r] * f + rs;
                fs[r] = f;
                s0[r] = p0; s1[r] = p1;
            }
#pragma unroll
            for (int t = 0; t < 8; ++t)
#pragma unroll
                for (int r = 0; r < 8; ++r) O[t][r] *= fs[r];

            // ---- P: C layout -> A layout via per-wave LDS staging ----
            _Float16* pw = plds[wv];
#pragma unroll
            for (int r = 0; r < 8; ++r) {
                int M = r + 8 * hi;
                pw[M * 32 + ln]      = (_Float16)s0[r];
                pw[M * 32 + 16 + ln] = (_Float16)s1[r];
            }
            const _Float16* prd = pw + ln * 32 + hi * 8;
            v16h aP = mk16(*(const v8h*)prd, *(const v8h*)(prd + 16));

            // ---- O += P V (8 hd sub-tiles) ----
#pragma unroll
            for (int t = 0; t < 8; ++t) {
#ifdef HAVE_TDM
                const _Float16* pv = vbl + hi * 8 * HDc + t * 16 + ln;
#else
                const _Float16* pv = vb + (size_t)(k0 + hi * 8) * HDc + t * 16 + ln;
#endif
                v16h bV;
#pragma unroll
                for (int i = 0; i < 8; ++i) { bV[i] = pv[i * HDc]; bV[i + 8] = pv[(i + 16) * HDc]; }
                O[t] = wmma_f16(aP, bV, O[t]);
            }
        }
#ifdef HAVE_TDM
        __syncthreads();    // all reads of buf[cur] done before overwrite
#endif
    }

    // ---- normalize + store to [B,N,inner] f16 ----
#pragma unroll
    for (int t = 0; t < 8; ++t) {
#pragma unroll
        for (int r = 0; r < 8; ++r) {
            int m = q0 + r + 8 * hi;
            float val = O[t][r] / lrun[r];
            oh[((size_t)b * Nc + m) * INNERc + h * HDc + t * 16 + ln] = (_Float16)val;
        }
    }
}

// ---------------------------------------------------------------------------
// Host-side launch
// ---------------------------------------------------------------------------
extern "C" void kernel_launch(void* const* d_in, const int* in_sizes, int n_in,
                              void* d_out, int out_size, void* d_ws, size_t ws_size,
                              hipStream_t stream) {
    (void)in_sizes; (void)n_in; (void)out_size; (void)ws_size;
    const float* x     = (const float*)d_in[0];   // [B,N,D]
    const float* w_qkv = (const float*)d_in[1];   // [3*inner, D]
    const float* w_o   = (const float*)d_in[2];   // [D, inner]
    float* out = (float*)d_out;                   // [B,N,D]

    char* ws = (char*)d_ws;
    const size_t SZ_XH    = (size_t)Bc * Nc * Dc * 2;
    const size_t SZ_WQKVT = (size_t)TRIPLEc * Dc * 2;
    const size_t SZ_WOT   = (size_t)Dc * INNERc * 2;
    const size_t SZ_HEAD  = (size_t)Bc * Hc * Nc * HDc * 2;

    _Float16* xh    = (_Float16*)(ws);
    _Float16* wqkvt = (_Float16*)(ws + SZ_XH);
    _Float16* wot   = (_Float16*)(ws + SZ_XH + SZ_WQKVT);
    _Float16* qraw  = (_Float16*)(ws + SZ_XH + SZ_WQKVT + SZ_WOT);
    _Float16* kraw  = (_Float16*)(ws + SZ_XH + SZ_WQKVT + SZ_WOT + SZ_HEAD);
    _Float16* vh    = (_Float16*)(ws + SZ_XH + SZ_WQKVT + SZ_WOT + 2 * SZ_HEAD);
    _Float16* kt    = (_Float16*)(ws + SZ_XH + SZ_WQKVT + SZ_WOT + 3 * SZ_HEAD);
    _Float16* ohb   = (_Float16*)(ws + SZ_XH + SZ_WQKVT + SZ_WOT + 4 * SZ_HEAD);

    {
        int n4 = (Bc * Nc * Dc) / 4;
        cvt_f32_f16<<<(n4 + 255) / 256, 256, 0, stream>>>(x, xh, n4);
    }
    transpose_f32_f16<<<dim3(Dc / 32, TRIPLEc / 32), dim3(32, 8), 0, stream>>>(w_qkv, wqkvt, TRIPLEc, Dc);
    transpose_f32_f16<<<dim3(INNERc / 32, Dc / 32), dim3(32, 8), 0, stream>>>(w_o, wot, Dc, INNERc);
    gemm_qkv<<<dim3((Bc * Nc) / 64, TRIPLEc / 256), 256, 0, stream>>>(xh, wqkvt, qraw, kraw, vh);
    {
        int nthreads = Bc * Hc * Nc * (HDc / 2);
        rope_kernel<<<nthreads / 256, 256, 0, stream>>>(qraw, kraw, kt);
    }
    flash_attn<<<dim3(Nc / 64, Bc * Hc), 128, 0, stream>>>(qraw, kt, vh, ohb);
    gemm_out<<<dim3((Bc * Nc) / 64, Dc / 256), 256, 0, stream>>>(ohb, wot, out);
}
